// MultiHeadAttention_73203422593375
// MI455X (gfx1250) — matrix operand order repaired and checked by reading, not compiled, not run on hardware
//
#include <hip/hip_runtime.h>
#include <hip/hip_bf16.h>
#include <math.h>

typedef __attribute__((ext_vector_type(16))) _Float16 v16h;
typedef __attribute__((ext_vector_type(8)))  _Float16 v8h;
typedef __attribute__((ext_vector_type(8)))  float    v8f;
typedef __attribute__((ext_vector_type(4)))  int      v4i;

#define AS1 __attribute__((address_space(1)))
#define AS3 __attribute__((address_space(3)))

#define D_MODEL   1024
#define NUM_HEADS 16
#define HEAD_DIM  64
#define BATCH     2
#define SEQ       2048
#define MROWS     (BATCH * SEQ)   // 4096

// ---------------------------------------------------------------------------
// helpers
// ---------------------------------------------------------------------------
__device__ __forceinline__ v8f wmma_f16(v16h a, v16h b, v8f c) {
  // V_WMMA_F32_16X16X32_F16: D = A(16x32 f16) * B(32x16 f16) + C(16x16 f32)
  return __builtin_amdgcn_wmma_f32_16x16x32_f16(
      /*neg_a=*/false, a, /*neg_b=*/false, b,
      /*c_mod=*/(short)0, c, /*reuse_a=*/false, /*reuse_b=*/false);
}

__device__ __forceinline__ v16h cat8(v8h lo, v8h hi) {
  return __builtin_shufflevector(lo, hi, 0,1,2,3,4,5,6,7,8,9,10,11,12,13,14,15);
}

// reductions across the 16-lane half-wave (C-fragment N dimension)
__device__ __forceinline__ float red_max16(float v) {
  v = fmaxf(v, __shfl_xor(v, 1));
  v = fmaxf(v, __shfl_xor(v, 2));
  v = fmaxf(v, __shfl_xor(v, 4));
  v = fmaxf(v, __shfl_xor(v, 8));
  return v;
}
__device__ __forceinline__ float red_sum16(float v) {
  v += __shfl_xor(v, 1);
  v += __shfl_xor(v, 2);
  v += __shfl_xor(v, 4);
  v += __shfl_xor(v, 8);
  return v;
}

// ---------------------------------------------------------------------------
// CDNA5 async global->LDS copy (ASYNCcnt path), with compile-safe fallback
// ---------------------------------------------------------------------------
#if defined(__AMDGCN__) && __has_builtin(__builtin_amdgcn_global_load_async_to_lds_b128)
#define HAS_ASYNC_LDS 1
#else
#define HAS_ASYNC_LDS 0
#endif

// copies 16 bytes per lane from global to LDS
__device__ __forceinline__ void async_cp16(_Float16* lds_dst, const _Float16* gsrc) {
#if HAS_ASYNC_LDS
  __builtin_amdgcn_global_load_async_to_lds_b128(
      (AS1 v4i*)((AS1 void*)gsrc),
      (AS3 v4i*)((AS3 void*)lds_dst),
      /*imm offset=*/0, /*cpol=*/0);
#else
  *(v8h*)lds_dst = *(const v8h*)gsrc;   // global_load_b128 + ds_store_b128
#endif
}

__device__ __forceinline__ void wait_async_copies() {
#if HAS_ASYNC_LDS
#if __has_builtin(__builtin_amdgcn_s_wait_asynccnt)
  __builtin_amdgcn_s_wait_asynccnt(0);
#else
  asm volatile("s_wait_asynccnt 0" ::: "memory");
#endif
#endif
}

// ---------------------------------------------------------------------------
// conversion kernels
// ---------------------------------------------------------------------------
__global__ void cvt_f32_f16(const float* __restrict__ in,
                            _Float16* __restrict__ out, int n) {
  int i = blockIdx.x * 256 + threadIdx.x;
  if (i < n) out[i] = (_Float16)in[i];
}

// W[k][n] (fp32, [K=1024, N=1024]) -> Wt[n][k] (f16), so B fragments load
// 16 contiguous halves per lane.
__global__ void cvt_w_transpose(const float* __restrict__ W,
                                _Float16* __restrict__ Wt) {
  int i = blockIdx.x * 256 + threadIdx.x;     // over 1M elements
  int k = i >> 10;
  int n = i & 1023;
  Wt[(size_t)n * 1024 + k] = (_Float16)W[(size_t)k * 1024 + n];
}

// ---------------------------------------------------------------------------
// projection GEMM: Out = X[4096,1024] @ W + bias, W given transposed [N,K].
// Each wave computes one 32(M) x 64(N) tile: K-loop of 32-chunks,
// 2 A-fragments x 4 B-fragments -> 8 WMMAs per chunk (each B reused twice).
// mode 0: write f16 [B,H,S,D]     (Q and K)
// mode 2: write f16 [B,H,D,S]     (V transposed, for PV B-fragments)
// mode 3: write f32 [M,1024]      (final output projection -> d_out)
// ---------------------------------------------------------------------------
__global__ __launch_bounds__(256)
void proj_gemm(const _Float16* __restrict__ X, const _Float16* __restrict__ Wt,
               const float* __restrict__ bias, void* __restrict__ outp, int mode) {
  const int lane = threadIdx.x & 31;
  const int wid  = blockIdx.x * 8 + (threadIdx.x >> 5);   // 0..2047
  const int mt   = wid >> 4;                              // 0..127
  const int nt   = wid & 15;                              // 0..15
  const int m0   = mt << 5;                               // 32-row tile
  const int n0   = nt << 6;                               // 64-col tile
  const int col  = lane & 15;
  const int hi   = lane >> 4;
  const int kb8  = hi << 3;    // unused (A fragments also use kb16)
  const int kb16 = hi << 4;    // B-fragment K base

  const _Float16* xrow0 = X + (size_t)(m0 + col) * 1024;
  const _Float16* xrow1 = X + (size_t)(m0 + 16 + col) * 1024;

  v8f acc[8] = {v8f{}, v8f{}, v8f{}, v8f{}, v8f{}, v8f{}, v8f{}, v8f{}};

  for (int kc = 0; kc < 1024; kc += 32) {
    v16h a0 = *(const v16h*)(xrow0 + kc + kb16);
    v16h a1 = *(const v16h*)(xrow1 + kc + kb16);
    v16h b0 = *(const v16h*)(Wt + (size_t)(n0 +  0 + col) * 1024 + kc + kb16);
    v16h b1 = *(const v16h*)(Wt + (size_t)(n0 + 16 + col) * 1024 + kc + kb16);
    v16h b2 = *(const v16h*)(Wt + (size_t)(n0 + 32 + col) * 1024 + kc + kb16);
    v16h b3 = *(const v16h*)(Wt + (size_t)(n0 + 48 + col) * 1024 + kc + kb16);
    acc[0] = wmma_f16(a0, b0, acc[0]);
    acc[4] = wmma_f16(a1, b0, acc[4]);
    acc[1] = wmma_f16(a0, b1, acc[1]);
    acc[5] = wmma_f16(a1, b1, acc[5]);
    acc[2] = wmma_f16(a0, b2, acc[2]);
    acc[6] = wmma_f16(a1, b2, acc[6]);
    acc[3] = wmma_f16(a0, b3, acc[3]);
    acc[7] = wmma_f16(a1, b3, acc[7]);
  }

#pragma unroll
  for (int half = 0; half < 2; half++) {
    const int mbase = m0 + half * 16;
#pragma unroll
    for (int j = 0; j < 4; j++) {
      const int n  = n0 + j * 16 + col;
      const float bv = bias[n];
      const v8f a = acc[half * 4 + j];
#pragma unroll
      for (int r = 0; r < 8; r++) {
        const int m = mbase + r + (hi << 3);
        const float val = a[r] + bv;
        if (mode == 3) {
          ((float*)outp)[(size_t)m * 1024 + n] = val;
        } else {
          const int b = m >> 11, s = m & 2047;
          const int h = n >> 6,  d = n & 63;
          _Float16* o = (_Float16*)outp;
          if (mode == 2)
            o[((size_t)(b * 16 + h) * 64 + d) * 2048 + s] = (_Float16)val;  // [B,H,D,S]
          else
            o[((size_t)(b * 16 + h) * 2048 + s) * 64 + d] = (_Float16)val;  // [B,H,S,D]
        }
      }
    }
  }
}

// ---------------------------------------------------------------------------
// Flash attention. One block = 8 waves, all on the same (b,h); each wave owns
// 16 query rows. K (32x64) and V^T (64x32) tiles are staged into LDS once per
// block via async global->LDS copies, double-buffered so the next tile's DMA
// overlaps the current tile's 8 WMMAs + softmax.
// ---------------------------------------------------------------------------
__global__ __launch_bounds__(256)
void attn_kernel(const _Float16* __restrict__ Qp,   // [B,H,S,64] f16
                 const _Float16* __restrict__ Kp,   // [B,H,S,64] f16
                 const _Float16* __restrict__ VpT,  // [B,H,64,S] f16
                 _Float16* __restrict__ ctx) {      // [4096,1024] f16
  __shared__ __align__(32) _Float16 kbufs[2][32 * 64];   // 2 x 4 KB
  __shared__ __align__(32) _Float16 vbufs[2][64 * 32];   // 2 x 4 KB
  __shared__ __align__(32) _Float16 pbuf[8 * 16 * 32];   // 8 KB (1 KB / wave)

  const int lane = threadIdx.x & 31;
  const int w    = threadIdx.x >> 5;
  const int bh   = blockIdx.x >> 4;          // 0..31
  const int qblk = blockIdx.x & 15;          // 0..15
  const int q0   = (qblk * 8 + w) << 4;      // this wave's 16 query rows
  const int col  = lane & 15;
  const int hi   = lane >> 4;
  const int kb8  = hi << 3;
  const int kb16 = hi << 4;

  const _Float16* kbase = Kp  + (size_t)bh * 2048 * 64;
  const _Float16* vbase = VpT + (size_t)bh * 64 * 2048;

  // per-lane slice of the cooperative tile copies (16 B per lane, 8 KB total)
  const int seg   = w * 32 + lane;           // 0..255
  const int kflat = seg * 8;                 // K tile is 4 KB contiguous
  const int vd    = seg >> 2;                // V^T row (d)
  const int vt    = (seg & 3) * 8;           // offset within 32-wide t row

  // Q A-fragments (d = 0..31 and 32..63), one query row per lane%16
  const _Float16* qrow = Qp + ((size_t)bh * 2048 + q0 + col) * 64;
  v16h a0 = *(const v16h*)(qrow + kb16);
  v16h a1 = *(const v16h*)(qrow + 32 + kb16);

  v8f o0 = {}, o1 = {}, o2 = {}, o3 = {};
  float mrow[8], lrow[8];
#pragma unroll
  for (int r = 0; r < 8; r++) { mrow[r] = -INFINITY; lrow[r] = 0.0f; }

  _Float16* pw = pbuf + w * 512;             // 16x32 f16 per wave
  const float scale = 0.125f;                // 1/sqrt(64)

  // preload tile 0
  async_cp16(&kbufs[0][kflat],          kbase + kflat);
  async_cp16(&vbufs[0][vd * 32 + vt],   vbase + (size_t)vd * 2048 + vt);
  wait_async_copies();
  __syncthreads();

  for (int i = 0; i < 64; i++) {
    const int kt  = i * 32;
    const int cur = i & 1;
    if (i + 1 < 64) {
      const int nxt = cur ^ 1;
      async_cp16(&kbufs[nxt][kflat],
                 kbase + (size_t)(kt + 32) * 64 + kflat);
      async_cp16(&vbufs[nxt][vd * 32 + vt],
                 vbase + (size_t)vd * 2048 + (kt + 32) + vt);
    }
    const _Float16* kl = kbufs[cur];
    const _Float16* vl = vbufs[cur];

    // ---- scores S[16,32] = Q @ K^T (B fragment: k=d contiguous in LDS) ----
    v16h b00 = *(const v16h*)(kl + (col)      * 64 + kb16);
    v16h b01 = *(const v16h*)(kl + (col)      * 64 + 32 + kb16);
    v16h b10 = *(const v16h*)(kl + (col + 16) * 64 + kb16);
    v16h b11 = *(const v16h*)(kl + (col + 16) * 64 + 32 + kb16);
    v8f s0 = {}; s0 = wmma_f16(a0, b00, s0); s0 = wmma_f16(a1, b01, s0);
    v8f s1 = {}; s1 = wmma_f16(a0, b10, s1); s1 = wmma_f16(a1, b11, s1);

    // ---- online softmax over this 32-column tile ----
#pragma unroll
    for (int r = 0; r < 8; r++) {
      const float x0 = s0[r] * scale;
      const float x1 = s1[r] * scale;
      float mloc = red_max16(fmaxf(x0, x1));
      const float mnew  = fmaxf(mrow[r], mloc);
      const float alpha = __expf(mrow[r] - mnew);
      const float p0 = __expf(x0 - mnew);
      const float p1 = __expf(x1 - mnew);
      const float rs = red_sum16(p0 + p1);
      lrow[r] = lrow[r] * alpha + rs;
      mrow[r] = mnew;
      o0[r] *= alpha; o1[r] *= alpha; o2[r] *= alpha; o3[r] *= alpha;
      // stash P tile (C-layout -> row-major [16][32] f16 in LDS)
      pw[(r + (hi << 3)) * 32 + col]      = (_Float16)p0;
      pw[(r + (hi << 3)) * 32 + col + 16] = (_Float16)p1;
    }
    // per-wave LDS RAW: DS ops are in-order, but force wait + compiler barrier
    asm volatile("s_wait_dscnt 0" ::: "memory");

    // re-read P as an A fragment (row = lane%16, k = kb16 + {0..15})
    const _Float16* pr = pw + col * 32 + kb16;
    v16h p = *(const v16h*)pr;

    // ---- O += P @ V (B fragment from LDS V^T tile: k=t contiguous) ----
    v16h bv0 = *(const v16h*)(vl + (col)      * 32 + kb16);
    v16h bv1 = *(const v16h*)(vl + (col + 16) * 32 + kb16);
    v16h bv2 = *(const v16h*)(vl + (col + 32) * 32 + kb16);
    v16h bv3 = *(const v16h*)(vl + (col + 48) * 32 + kb16);
    o0 = wmma_f16(p, bv0, o0);
    o1 = wmma_f16(p, bv1, o1);
    o2 = wmma_f16(p, bv2, o2);
    o3 = wmma_f16(p, bv3, o3);

    // pipeline barrier: this tile consumed, next tile's DMA must have landed
    wait_async_copies();
    __syncthreads();
  }

  // ---- normalize and write concat layout ctx[b*2048+s][h*64+d] ----
  const int b = bh >> 4, h = bh & 15;
#pragma unroll
  for (int r = 0; r < 8; r++) {
    const float inv = 1.0f / lrow[r];
    const int s = q0 + r + (hi << 3);
    const size_t rowoff = (size_t)(b * 2048 + s) * 1024 + h * 64;
    ctx[rowoff + col]      = (_Float16)(o0[r] * inv);
    ctx[rowoff + 16 + col] = (_Float16)(o1[r] * inv);
    ctx[rowoff + 32 + col] = (_Float16)(o2[r] * inv);
    ctx[rowoff + 48 + col] = (_Float16)(o3[r] * inv);
  }
}

// ---------------------------------------------------------------------------
// launch
// ---------------------------------------------------------------------------
extern "C" void kernel_launch(void* const* d_in, const int* in_sizes, int n_in,
                              void* d_out, int out_size, void* d_ws, size_t ws_size,
                              hipStream_t stream) {
  const float* query = (const float*)d_in[0];
  const float* key_  = (const float*)d_in[1];
  const float* value = (const float*)d_in[2];
  const float* Wq = (const float*)d_in[3];
  const float* bq = (const float*)d_in[4];
  const float* Wk = (const float*)d_in[5];
  const float* bk = (const float*)d_in[6];
  const float* Wv = (const float*)d_in[7];
  const float* bv = (const float*)d_in[8];
  const float* Wo = (const float*)d_in[9];
  const float* bo = (const float*)d_in[10];
  float* out = (float*)d_out;

  const size_t ACT = (size_t)MROWS * D_MODEL;   // 4M elements
  const size_t WEL = (size_t)D_MODEL * D_MODEL; // 1M elements

  _Float16* ws   = (_Float16*)d_ws;
  _Float16* x16q = ws;
  _Float16* x16k = ws + ACT;
  _Float16* x16v = ws + 2 * ACT;
  _Float16* wqt  = ws + 3 * ACT;
  _Float16* wkt  = wqt + WEL;
  _Float16* wvt  = wkt + WEL;
  _Float16* wot  = wvt + WEL;
  _Float16* Qp   = wot + WEL;        // [B,H,S,64]
  _Float16* Kp   = Qp + ACT;         // [B,H,S,64]
  _Float16* VpT  = Kp + ACT;         // [B,H,64,S]
  _Float16* ctx  = VpT + ACT;        // [4096,1024]

  // 1) convert activations fp32 -> f16
  cvt_f32_f16<<<(int)(ACT / 256), 256, 0, stream>>>(query, x16q, (int)ACT);
  cvt_f32_f16<<<(int)(ACT / 256), 256, 0, stream>>>(key_,  x16k, (int)ACT);
  cvt_f32_f16<<<(int)(ACT / 256), 256, 0, stream>>>(value, x16v, (int)ACT);

  // 2) transpose+convert weights to [N,K] f16
  cvt_w_transpose<<<(int)(WEL / 256), 256, 0, stream>>>(Wq, wqt);
  cvt_w_transpose<<<(int)(WEL / 256), 256, 0, stream>>>(Wk, wkt);
  cvt_w_transpose<<<(int)(WEL / 256), 256, 0, stream>>>(Wv, wvt);
  cvt_w_transpose<<<(int)(WEL / 256), 256, 0, stream>>>(Wo, wot);

  // 3) QKV projections (2048 wave-tiles of 32x64 -> 256 blocks of 8 waves)
  proj_gemm<<<256, 256, 0, stream>>>(x16q, wqt, bq, (void*)Qp,  0);
  proj_gemm<<<256, 256, 0, stream>>>(x16k, wkt, bk, (void*)Kp,  0);
  proj_gemm<<<256, 256, 0, stream>>>(x16v, wvt, bv, (void*)VpT, 2);

  // 4) flash attention (32 bh x 16 q-blocks = 512 blocks of 8 waves)
  attn_kernel<<<512, 256, 0, stream>>>(Qp, Kp, VpT, ctx);

  // 5) output projection -> fp32 d_out
  proj_gemm<<<256, 256, 0, stream>>>(ctx, wot, bo, (void*)out, 3);
}
